// MultiHeadMinkUnet_87814901334081
// MI455X (gfx1250) — matrix operand order repaired and checked by reading, not compiled, not run on hardware
//
#include <hip/hip_runtime.h>
#include <hip/hip_bf16.h>

#define NPTS 65536
#define KNBR 27
#define LOGP 4.61512051684126f
#define EPSBN 1e-5f

typedef __attribute__((ext_vector_type(16))) _Float16 v16h;
typedef __attribute__((ext_vector_type(8)))  _Float16 v8h;
typedef __attribute__((ext_vector_type(8)))  float    v8f;

// ---- WMMA fragment loaders (contiguous k order shared by A and B, wave32) ----

// A-matrix 16x32 f16: lane row = lane&15; lanes0-15 hold K={kb..kb+15},
// lanes16-31 hold K={kb+16..kb+31} (contiguous, matches B).
static __device__ __forceinline__ v16h frag_a_lds(const _Float16* s, int stride, int kb, int lane) {
  const int row = lane & 15;
  const int hi  = lane >> 4;
  const _Float16* p0 = s + row * stride + kb + (hi << 4);
  v8h lo = *(const v8h*)p0;          // K = kb + 16*hi + 0..7
  v8h hh = *(const v8h*)(p0 + 8);    // K = kb + 16*hi + 8..15
  v16h a;
#pragma unroll
  for (int i = 0; i < 8; ++i) { a[i] = lo[i]; a[8 + i] = hh[i]; }
  return a;
}

// B-matrix 32x16 f16: lane col = lane&15; 16 contiguous K starting at kb + 16*(lane>>4).
// wT is column-major: wT[col*Kdim + k]  ->  one 32B contiguous load per lane.
static __device__ __forceinline__ v16h frag_b_glb(const _Float16* __restrict__ wT, int Kdim,
                                                  int colbase, int kb, int lane) {
  const int col = colbase + (lane & 15);
  const int ks  = kb + ((lane >> 4) << 4);
  return *(const v16h*)(wT + (size_t)col * Kdim + ks);
}

static __device__ __forceinline__ v8f wmma_f16(v16h a, v16h b, v8f c) {
  return __builtin_amdgcn_wmma_f32_16x16x32_f16(false, a, false, b, (short)0, c, false, false);
}

// ---- 0: pack all GEMM weights into f16 column-major (B-fragment friendly) ----
__global__ __launch_bounds__(256) void pack_weights_kernel(
    const float* __restrict__ w_e1, const float* __restrict__ w_e2,
    const float* __restrict__ w_lab, const float* __restrict__ w_unlab,
    const float* __restrict__ w_over, const float* __restrict__ w_clip1,
    const float* __restrict__ w_clip2,
    _Float16* __restrict__ wTe1, _Float16* __restrict__ wTe2,
    _Float16* __restrict__ wTh,  _Float16* __restrict__ wTc1,
    _Float16* __restrict__ wTc2) {
  int i = blockIdx.x * 256 + threadIdx.x;
  if (i < 96 * 1728) { int d = i / 1728, t = i % 1728; wTe1[i] = (_Float16)w_e1[(size_t)t * 96 + d]; return; }
  i -= 96 * 1728;
  if (i < 96 * 2592) { int d = i / 2592, t = i % 2592; wTe2[i] = (_Float16)w_e2[(size_t)t * 96 + d]; return; }
  i -= 96 * 2592;
  if (i < 128 * 96) {
    int d = i / 96, c = i % 96;
    float v = 0.0f;
    if (d < 19)       v = w_lab[c * 19 + d];
    else if (d < 44)  { int q = d - 19; v = w_unlab[((q / 5) * 96 + c) * 5 + (q % 5)]; }
    else if (d < 119) { int q = d - 44; v = w_over[((q / 15) * 96 + c) * 15 + (q % 15)]; }
    wTh[i] = (_Float16)v; return;
  }
  i -= 128 * 96;
  if (i < 256 * 96)  { int d = i / 96,  c = i % 96;  wTc1[i] = (_Float16)w_clip1[c * 256 + d]; return; }
  i -= 256 * 96;
  if (i < 512 * 256) { int d = i / 256, c = i % 256; wTc2[i] = (_Float16)w_clip2[c * 512 + d]; return; }
}

// ---- 1: per-node entropy gate + rgb(4->32) + img(101->32), emit x0 f16 (N,64) ----
__global__ __launch_bounds__(256) void encode_kernel(
    const float* __restrict__ feats,
    const float* __restrict__ w_rgb, const float* __restrict__ b_rgb,
    const float* __restrict__ bn_rgb,
    const float* __restrict__ w_img, const float* __restrict__ b_img,
    const float* __restrict__ bn_img,
    _Float16* __restrict__ x0) {
  __shared__ float swi[101 * 32];
  __shared__ float swr[4 * 32];
  const int tid = threadIdx.x;
  for (int i = tid; i < 101 * 32; i += 256) swi[i] = w_img[i];
  if (tid < 128) swr[tid] = w_rgb[tid];
  __syncthreads();

  const int n = blockIdx.x * 256 + tid;
  const float* f = feats + (size_t)n * 105;

  float H = 0.0f;
  for (int i = 0; i < 101; ++i) {
    float ps = fmaxf(f[4 + i], 1e-12f);
    H -= ps * __logf(ps);
  }
  const float wgt = 1.0f - H / LOGP;

  float acc[32];
#pragma unroll
  for (int j = 0; j < 32; ++j) acc[j] = 0.0f;
  for (int i = 0; i < 101; ++i) {
    float pv = f[4 + i];
#pragma unroll
    for (int j = 0; j < 32; ++j) acc[j] = fmaf(pv, swi[i * 32 + j], acc[j]);
  }
  const float r0 = f[0], r1 = f[1], r2 = f[2], r3 = f[3];

  _Float16* row = x0 + (size_t)n * 64;
#pragma unroll 8
  for (int j = 0; j < 32; ++j) {
    float t = r0 * swr[j] + r1 * swr[32 + j] + r2 * swr[64 + j] + r3 * swr[96 + j] + b_rgb[j];
    float s = bn_rgb[j] * rsqrtf(bn_rgb[96 + j] + EPSBN);
    row[j] = (_Float16)fmaxf(s * (t - bn_rgb[64 + j]) + bn_rgb[32 + j], 0.0f);
    float u = wgt * acc[j] + b_img[j];
    float si = bn_img[j] * rsqrtf(bn_img[96 + j] + EPSBN);
    row[32 + j] = (_Float16)fmaxf(si * (u - bn_img[64 + j]) + bn_img[32 + j], 0.0f);
  }
}

// ---- 2/3: sparse conv: out[n,d] = relu(bn( sum_k sum_c x[nbr[n,k],c] * W[k,c,d] ))
// 32 nodes/block, 6 waves (one 16-col WMMA tile, two 16-row tiles sharing B),
// double-buffered LDS gather pipeline: global loads for k+1 overlap WMMAs of k.
template <int CIN, bool WRITE_F32>
__global__ __launch_bounds__(192) void spconv_kernel(
    const _Float16* __restrict__ xin, const int* __restrict__ nbr,
    const _Float16* __restrict__ wT,  const float* __restrict__ bnp,
    _Float16* __restrict__ outh, float* __restrict__ outf) {
  constexpr int KD     = KNBR * CIN;     // full K dimension
  constexpr int CPR    = CIN / 8;        // 16B chunks per row
  constexpr int NCHUNK = 32 * CPR;       // chunks per k-slice (32 rows)
  constexpr int STEPS  = CIN / 32;       // WMMA K-steps per neighbor
  __shared__ __align__(32) _Float16 sA[2][32 * CIN];

  const int tid = threadIdx.x;
  const int wave = tid >> 5;
  const int lane = tid & 31;
  const int nodebase = blockIdx.x << 5;  // 32 nodes per block
  const int colbase = wave << 4;

  // per-thread gather slots (up to 2 chunks each)
  int  crow[2], ccol[2];
  bool cval[2];
#pragma unroll
  for (int j = 0; j < 2; ++j) {
    int idx = tid + j * 192;
    cval[j] = idx < NCHUNK;
    crow[j] = idx / CPR;
    ccol[j] = (idx % CPR) << 3;
  }

  // per-lane B pointer: advances by compile-time constants only
  const _Float16* bptr = wT + (size_t)(colbase + (lane & 15)) * KD + ((lane >> 4) << 4);

  // prefetch gather for k = 0 into registers
  v8h stage[2];
#pragma unroll
  for (int j = 0; j < 2; ++j) {
    if (cval[j]) {
      int nb = nbr[(nodebase + crow[j]) * KNBR];
      stage[j] = *(const v8h*)(xin + (size_t)nb * CIN + ccol[j]);
    }
  }

  v8f acc0 = {}, acc1 = {};
  for (int k = 0; k < KNBR; ++k) {
    _Float16* buf = sA[k & 1];
    // commit staged rows for this k
#pragma unroll
    for (int j = 0; j < 2; ++j)
      if (cval[j]) *(v8h*)(buf + crow[j] * CIN + ccol[j]) = stage[j];
    __syncthreads();
    // issue gather for k+1 (in flight during the WMMAs below)
    if (k + 1 < KNBR) {
#pragma unroll
      for (int j = 0; j < 2; ++j) {
        if (cval[j]) {
          int nb = nbr[(nodebase + crow[j]) * KNBR + (k + 1)];
          stage[j] = *(const v8h*)(xin + (size_t)nb * CIN + ccol[j]);
        }
      }
    }
#pragma unroll
    for (int s = 0; s < STEPS; ++s) {
      v16h b  = *(const v16h*)(bptr + k * CIN + s * 32);
      v16h a0 = frag_a_lds(buf, CIN, s * 32, lane);
      acc0 = wmma_f16(a0, b, acc0);
      v16h a1 = frag_a_lds(buf + 16 * CIN, CIN, s * 32, lane);
      acc1 = wmma_f16(a1, b, acc1);
    }
    // no trailing barrier: buf is rewritten only at iteration k+2, and the
    // k+1 barrier already orders those writes after this iteration's reads.
  }

  const int col = colbase + (lane & 15);
  const int hi  = lane >> 4;
  const float sc = bnp[col] * rsqrtf(bnp[288 + col] + EPSBN);
  const float bb = bnp[96 + col], mm = bnp[192 + col];
#pragma unroll
  for (int t = 0; t < 2; ++t) {
    const v8f& a = t ? acc1 : acc0;
#pragma unroll
    for (int r = 0; r < 8; ++r) {
      int row = nodebase + (t << 4) + r + (hi << 3);
      float y = fmaxf(sc * (a[r] - mm) + bb, 0.0f);
      outh[(size_t)row * 96 + col] = (_Float16)y;
      if (WRITE_F32) outf[(size_t)row * 96 + col] = y;
    }
  }
}

// ---- 4: classification heads (96 -> 19 | 5x5 | 5x15), padded to 128 cols ----
__global__ __launch_bounds__(256) void heads_kernel(
    const _Float16* __restrict__ x2h, const _Float16* __restrict__ wTh,
    float* __restrict__ out_lab, float* __restrict__ out_unlab,
    float* __restrict__ out_over) {
  __shared__ __align__(32) _Float16 sA[16 * 96];
  const int tid = threadIdx.x;
  const int wave = tid >> 5;
  const int lane = tid & 31;
  const int nodebase = blockIdx.x << 4;
  for (int idx = tid; idx < 192; idx += 256) {
    int r = idx / 12, c = (idx % 12) << 3;
    *(v8h*)(sA + r * 96 + c) = *(const v8h*)(x2h + (size_t)(nodebase + r) * 96 + c);
  }
  __syncthreads();
  const int colbase = wave << 4;
  v8f acc = {};
#pragma unroll
  for (int s = 0; s < 3; ++s) {
    v16h a = frag_a_lds(sA, 96, s << 5, lane);
    v16h b = frag_b_glb(wTh, 96, colbase, s << 5, lane);
    acc = wmma_f16(a, b, acc);
  }
  // resolve scatter target ONCE per lane (col invariant across the 8 rows)
  const int col = colbase + (lane & 15);
  const int hi  = lane >> 4;
  float* dst = nullptr;
  int stride = 0;
  if (col < 19)       { dst = out_lab + col; stride = 19; }
  else if (col < 44)  { int q = col - 19; dst = out_unlab + (size_t)(q / 5) * NPTS * 5 + (q % 5); stride = 5; }
  else if (col < 119) { int q = col - 44; dst = out_over + (size_t)(q / 15) * NPTS * 15 + (q % 15); stride = 15; }
  if (dst) {
    dst += (size_t)(nodebase + (hi << 3)) * stride;
#pragma unroll
    for (int r = 0; r < 8; ++r) dst[(size_t)r * stride] = acc[r];
  }
}

// ---- 5: clip1: relu(bn(x @ W(96x256) + b)) -> h f16 ----
__global__ __launch_bounds__(256) void clip1_kernel(
    const _Float16* __restrict__ x2h, const _Float16* __restrict__ wTc1,
    const float* __restrict__ bias, const float* __restrict__ bnp,
    _Float16* __restrict__ hh) {
  __shared__ __align__(32) _Float16 sA[16 * 96];
  const int tid = threadIdx.x;
  const int wave = tid >> 5;
  const int lane = tid & 31;
  const int nodebase = blockIdx.x << 4;
  for (int idx = tid; idx < 192; idx += 256) {
    int r = idx / 12, c = (idx % 12) << 3;
    *(v8h*)(sA + r * 96 + c) = *(const v8h*)(x2h + (size_t)(nodebase + r) * 96 + c);
  }
  __syncthreads();
  const int cb0 = wave << 5;           // wave*32: two 16-col tiles per wave
  v8f acc0 = {}, acc1 = {};
#pragma unroll
  for (int s = 0; s < 3; ++s) {
    v16h a  = frag_a_lds(sA, 96, s << 5, lane);
    v16h b0 = frag_b_glb(wTc1, 96, cb0,      s << 5, lane);
    acc0 = wmma_f16(a, b0, acc0);
    v16h b1 = frag_b_glb(wTc1, 96, cb0 + 16, s << 5, lane);
    acc1 = wmma_f16(a, b1, acc1);
  }
  const int hi = lane >> 4;
#pragma unroll
  for (int t = 0; t < 2; ++t) {
    v8f* acc = t ? &acc1 : &acc0;
    int col = cb0 + (t << 4) + (lane & 15);
    float sc = bnp[col] * rsqrtf(bnp[768 + col] + EPSBN);
    float bb = bnp[256 + col], mm = bnp[512 + col], bv = bias[col];
#pragma unroll
    for (int r = 0; r < 8; ++r) {
      int row = nodebase + r + (hi << 3);
      float y = fmaxf(sc * ((*acc)[r] + bv - mm) + bb, 0.0f);
      hh[(size_t)row * 256 + col] = (_Float16)y;
    }
  }
}

// ---- 6: clip2: h @ W(256x512) + b -> clip_feats f32 ----
__global__ __launch_bounds__(256) void clip2_kernel(
    const _Float16* __restrict__ hh, const _Float16* __restrict__ wTc2,
    const float* __restrict__ bias, float* __restrict__ outc) {
  __shared__ __align__(32) _Float16 sA[16 * 256];
  const int tid = threadIdx.x;
  const int wave = tid >> 5;
  const int lane = tid & 31;
  const int nodebase = blockIdx.x << 4;
  for (int idx = tid; idx < 512; idx += 256) {
    int r = idx >> 5, c = (idx & 31) << 3;
    *(v8h*)(sA + r * 256 + c) = *(const v8h*)(hh + (size_t)(nodebase + r) * 256 + c);
  }
  __syncthreads();
  const int cbase = wave << 6;         // wave*64: four 16-col tiles per wave
  v8f acc[4] = {{}, {}, {}, {}};
  for (int s = 0; s < 8; ++s) {
    v16h a = frag_a_lds(sA, 256, s << 5, lane);
#pragma unroll
    for (int t = 0; t < 4; ++t) {
      v16h b = frag_b_glb(wTc2, 256, cbase + (t << 4), s << 5, lane);
      acc[t] = wmma_f16(a, b, acc[t]);
    }
  }
  const int hi = lane >> 4;
#pragma unroll
  for (int t = 0; t < 4; ++t) {
    int col = cbase + (t << 4) + (lane & 15);
    float bv = bias[col];
#pragma unroll
    for (int r = 0; r < 8; ++r) {
      int row = nodebase + r + (hi << 3);
      outc[(size_t)row * 512 + col] = acc[t][r] + bv;
    }
  }
}

extern "C" void kernel_launch(void* const* d_in, const int* in_sizes, int n_in,
                              void* d_out, int out_size, void* d_ws, size_t ws_size,
                              hipStream_t stream) {
  const float* feats   = (const float*)d_in[0];
  const int*   nbr1    = (const int*)d_in[1];
  const int*   nbr2    = (const int*)d_in[2];
  const float* w_rgb   = (const float*)d_in[3];
  const float* b_rgb   = (const float*)d_in[4];
  const float* bn_rgb  = (const float*)d_in[5];
  const float* w_img   = (const float*)d_in[6];
  const float* b_img   = (const float*)d_in[7];
  const float* bn_img  = (const float*)d_in[8];
  const float* w_e1    = (const float*)d_in[9];
  const float* w_e2    = (const float*)d_in[10];
  const float* bn_e1   = (const float*)d_in[11];
  const float* bn_e2   = (const float*)d_in[12];
  const float* w_lab   = (const float*)d_in[13];
  const float* w_unlab = (const float*)d_in[14];
  const float* w_over  = (const float*)d_in[15];
  const float* w_clip1 = (const float*)d_in[16];
  const float* b_clip1 = (const float*)d_in[17];
  const float* bn_clip = (const float*)d_in[18];
  const float* w_clip2 = (const float*)d_in[19];
  const float* b_clip2 = (const float*)d_in[20];

  // workspace layout (all 256B-aligned)
  char* ws = (char*)d_ws;
  const size_t OFF_X0  = 0;
  const size_t OFF_X1  = OFF_X0  + (size_t)NPTS * 64  * 2;
  const size_t OFF_X2  = OFF_X1  + (size_t)NPTS * 96  * 2;
  const size_t OFF_H   = OFF_X2  + (size_t)NPTS * 96  * 2;
  const size_t OFF_WE1 = OFF_H   + (size_t)NPTS * 256 * 2;
  const size_t OFF_WE2 = OFF_WE1 + (size_t)1728 * 96  * 2;
  const size_t OFF_WH  = OFF_WE2 + (size_t)2592 * 96  * 2;
  const size_t OFF_WC1 = OFF_WH  + (size_t)128  * 96  * 2;
  const size_t OFF_WC2 = OFF_WC1 + (size_t)256  * 96  * 2;

  _Float16* x0   = (_Float16*)(ws + OFF_X0);
  _Float16* x1   = (_Float16*)(ws + OFF_X1);
  _Float16* x2   = (_Float16*)(ws + OFF_X2);
  _Float16* hbuf = (_Float16*)(ws + OFF_H);
  _Float16* wTe1 = (_Float16*)(ws + OFF_WE1);
  _Float16* wTe2 = (_Float16*)(ws + OFF_WE2);
  _Float16* wTh  = (_Float16*)(ws + OFF_WH);
  _Float16* wTc1 = (_Float16*)(ws + OFF_WC1);
  _Float16* wTc2 = (_Float16*)(ws + OFF_WC2);

  // output regions (tuple flattened in return order)
  float* out       = (float*)d_out;
  float* out_lab   = out;
  float* out_unlab = out_lab   + (size_t)NPTS * 19;
  float* out_over  = out_unlab + (size_t)5 * NPTS * 5;
  float* out_x     = out_over  + (size_t)5 * NPTS * 15;
  float* out_clip  = out_x     + (size_t)NPTS * 96;

  const int packN = 96 * 1728 + 96 * 2592 + 128 * 96 + 256 * 96 + 512 * 256;
  pack_weights_kernel<<<(packN + 255) / 256, 256, 0, stream>>>(
      w_e1, w_e2, w_lab, w_unlab, w_over, w_clip1, w_clip2,
      wTe1, wTe2, wTh, wTc1, wTc2);

  encode_kernel<<<NPTS / 256, 256, 0, stream>>>(
      feats, w_rgb, b_rgb, bn_rgb, w_img, b_img, bn_img, x0);

  spconv_kernel<64, false><<<NPTS / 32, 192, 0, stream>>>(x0, nbr1, wTe1, bn_e1, x1, nullptr);
  spconv_kernel<96, true><<<NPTS / 32, 192, 0, stream>>>(x1, nbr2, wTe2, bn_e2, x2, out_x);

  heads_kernel<<<NPTS / 16, 256, 0, stream>>>(x2, wTh, out_lab, out_unlab, out_over);
  clip1_kernel<<<NPTS / 16, 256, 0, stream>>>(x2, wTc1, b_clip1, bn_clip, hbuf);
  clip2_kernel<<<NPTS / 16, 256, 0, stream>>>(hbuf, wTc2, b_clip2, out_clip);
}
